// GAT_63299228009394
// MI455X (gfx1250) — matrix operand order repaired and checked by reading, not compiled, not run on hardware
//
#include <hip/hip_runtime.h>
#include <hip/hip_bf16.h>
#include <math.h>

#define HEADS 8
#define EPS_BN 1e-5f
#define NEG_SLOPE 0.2f

typedef __attribute__((ext_vector_type(16))) __bf16         v16bf;
typedef __attribute__((ext_vector_type(16))) unsigned short v16u;
typedef __attribute__((ext_vector_type(8)))  unsigned short v8u;
typedef __attribute__((ext_vector_type(8)))  float          v8f;

// ---------- helpers ----------
__device__ __forceinline__ unsigned short f2bf(float f) {
    unsigned int u = __float_as_uint(f);
    u += 0x7FFFu + ((u >> 16) & 1u);          // round-to-nearest-even
    return (unsigned short)(u >> 16);
}

__device__ __forceinline__ float lrelu(float x) { return x > 0.f ? x : NEG_SLOPE * x; }

// float atomic max via signed/unsigned int punning (buffer initialized to -inf)
__device__ __forceinline__ void atomicMaxF(float* addr, float val) {
    if (val >= 0.f) atomicMax((int*)addr, __float_as_int(val));
    else            atomicMin((unsigned int*)addr, __float_as_uint(val));
}

// ---------- generic fill ----------
__global__ void fill_f32(float* p, int n, float v) {
    int t = blockIdx.x * blockDim.x + threadIdx.x;
    if (t < n) p[t] = v;
}

// ---------- x: f32 -> bf16 with zero column padding (row-major A matrix) ----------
__global__ void cast_pad_bf16(const float* __restrict__ in, unsigned short* __restrict__ out,
                              int Rout, int Cout_, int Rin, int Cin) {
    long t = (long)blockIdx.x * blockDim.x + threadIdx.x;
    long tot = (long)Rout * Cout_;
    if (t >= tot) return;
    int r = (int)(t / Cout_), c = (int)(t % Cout_);
    float v = (r < Rin && c < Cin) ? in[(long)r * Cin + c] : 0.f;
    out[t] = f2bf(v);
}

// ---------- weights: f32 [Kin,Nc] -> bf16 packed in WMMA B-fragment order ----------
// Packed layout: [kt][nt][lane 0..31][16 bf16] so a lane's whole fragment is one
// contiguous 32-byte load. Fragment map (ISA 7.12.2, 16-bit B 32x16, wave32):
//   lane n = lane%16, K-base = 8*(lane/16); elem i -> K = kb + (i&7) + 16*(i>>3).
__global__ void pack_b_frag(const float* __restrict__ W, unsigned short* __restrict__ out,
                            int Kpad, int Nc, int Kin) {
    long t = (long)blockIdx.x * blockDim.x + threadIdx.x;
    long tot = (long)Kpad * Nc;
    if (t >= tot) return;
    int i    = (int)(t & 15);
    int lane = (int)((t >> 4) & 31);
    long fr  = t >> 9;                      // fragment index = kt*ntN + nt
    int ntN  = Nc >> 4;
    int nt   = (int)(fr % ntN);
    int kt   = (int)(fr / ntN);
    int n    = lane & 15;
    int kb   = (lane >> 4) << 3;            // 0 or 8
    int k    = (kt << 5) + kb + (i & 7) + ((i >> 3) << 4);
    int col  = (nt << 4) + n;
    float v  = (k < Kin) ? W[(long)k * Nc + col] : 0.f;
    out[t] = f2bf(v);
}

// ---------- WMMA GEMM: C[M,Nc] = A[M,K](bf16 row-major) @ B(packed frags) ----------
// Each wave computes a 16x64 strip: one A fragment feeds 4 WMMAs per K-step.
// A block's 8 waves span 8/ng m-tiles; waves with the same mt share A rows (A reuse in L0).
__global__ __launch_bounds__(256) void gat_gemm_wmma(
        const unsigned short* __restrict__ A, const unsigned short* __restrict__ Bp,
        float* __restrict__ C, int M, int K, int Nc) {
    const int lane = threadIdx.x & 31;
    const int wid  = threadIdx.x >> 5;
    const int ntN  = Nc >> 4;
    const int ng   = ntN >> 2;              // n-tile groups of 4
    const int ntM  = M >> 4;
    const int wt   = blockIdx.x * (blockDim.x >> 5) + wid;
    if (wt >= ntM * ng) return;             // wave-uniform: EXEC stays all-1s
    const int mt  = wt / ng;
    const int nt0 = (wt % ng) << 2;

    const int am = (mt << 4) + (lane & 15);
    const int kh = (lane >> 4) << 3;        // 0 or 8

    v8f acc0 = {}, acc1 = {}, acc2 = {}, acc3 = {};
    const int ktN = K >> 5;
    const long FR = 32 * 16;                // shorts per fragment

    for (int kt = 0; kt < ktN; ++kt) {
        const unsigned short* ap = A + (long)am * K + (kt << 5) + kh;
        v8u alo = *(const v8u*)ap;
        v8u ahi = *(const v8u*)(ap + 16);
        v16bf a = __builtin_bit_cast(v16bf,
            __builtin_shufflevector(alo, ahi, 0,1,2,3,4,5,6,7,8,9,10,11,12,13,14,15));

        const unsigned short* bb =
            Bp + (((long)kt * ntN + nt0) * 32 + lane) * 16;
        v16bf b0 = __builtin_bit_cast(v16bf, *(const v16u*)(bb));
        v16bf b1 = __builtin_bit_cast(v16bf, *(const v16u*)(bb + FR));
        v16bf b2 = __builtin_bit_cast(v16bf, *(const v16u*)(bb + 2 * FR));
        v16bf b3 = __builtin_bit_cast(v16bf, *(const v16u*)(bb + 3 * FR));

        if (kt + 1 < ktN) {                 // global_prefetch_b8 next K-step
            __builtin_prefetch(ap + 32, 0, 1);
            __builtin_prefetch(bb + (long)ntN * FR, 0, 1);
        }

        acc0 = __builtin_amdgcn_wmma_f32_16x16x32_bf16(false, a, false, b0, (short)0, acc0, false, false);
        acc1 = __builtin_amdgcn_wmma_f32_16x16x32_bf16(false, a, false, b1, (short)0, acc1, false, false);
        acc2 = __builtin_amdgcn_wmma_f32_16x16x32_bf16(false, a, false, b2, (short)0, acc2, false, false);
        acc3 = __builtin_amdgcn_wmma_f32_16x16x32_bf16(false, a, false, b3, (short)0, acc3, false, false);
    }

    const int cm  = (mt << 4) + ((lane >> 4) << 3);
    const int cn0 = (nt0 << 4) + (lane & 15);
    float* c0 = C + (long)cm * Nc + cn0;
#pragma unroll
    for (int r = 0; r < 8; ++r) {
        float* cr = c0 + (long)r * Nc;
        cr[0]  = acc0[r];
        cr[16] = acc1[r];
        cr[32] = acc2[r];
        cr[48] = acc3[r];
    }
}

// ---------- attention logits: al[n,h] = sum_c h[n,h,c] * a[h,c] ----------
__global__ void attn_logits(const float* __restrict__ h, const float* __restrict__ a_s,
                            const float* __restrict__ a_d, float* __restrict__ als,
                            float* __restrict__ ald, int n, int C) {
    int t = blockIdx.x * blockDim.x + threadIdx.x;
    if (t >= n * HEADS) return;
    int node = t >> 3, hd = t & 7;
    const float* hp = h + (long)node * (HEADS * C) + hd * C;
    const float* sp = a_s + hd * C;
    const float* dp = a_d + hd * C;
    float s = 0.f, d = 0.f;
    for (int c = 0; c < C; ++c) { s += hp[c] * sp[c]; d += hp[c] * dp[c]; }
    als[t] = s; ald[t] = d;
}

// ---------- edge softmax: pass 1 (segment max over dst) ----------
__global__ void edge_max(const int* __restrict__ src, const int* __restrict__ dst, int E, int TE,
                         const float* __restrict__ als, const float* __restrict__ ald,
                         float* __restrict__ mbuf) {
    int t = blockIdx.x * blockDim.x + threadIdx.x;
    if (t >= TE * HEADS) return;
    int e = t >> 3, hd = t & 7;
    int s, d;
    if (e < E) { s = src[e]; d = dst[e]; } else { s = d = e - E; }   // self loops
    float v = lrelu(als[s * HEADS + hd] + ald[d * HEADS + hd]);
    atomicMaxF(&mbuf[d * HEADS + hd], v);
}

// ---------- edge softmax: pass 2 (exp-sum) ----------
__global__ void edge_sum(const int* __restrict__ src, const int* __restrict__ dst, int E, int TE,
                         const float* __restrict__ als, const float* __restrict__ ald,
                         const float* __restrict__ mbuf, float* __restrict__ sbuf) {
    int t = blockIdx.x * blockDim.x + threadIdx.x;
    if (t >= TE * HEADS) return;
    int e = t >> 3, hd = t & 7;
    int s, d;
    if (e < E) { s = src[e]; d = dst[e]; } else { s = d = e - E; }
    float v = lrelu(als[s * HEADS + hd] + ald[d * HEADS + hd]);
    float ex = __expf(v - mbuf[d * HEADS + hd]);
    atomicAdd(&sbuf[d * HEADS + hd], ex);
}

// ---------- edge scatter: out[dst] += h[src] * alpha ----------
__global__ void edge_scatter(const int* __restrict__ src, const int* __restrict__ dst, int E, int TE,
                             const float* __restrict__ als, const float* __restrict__ ald,
                             const float* __restrict__ mbuf, const float* __restrict__ sbuf,
                             const float* __restrict__ h, float* __restrict__ obuf,
                             int Cout_, int C) {
    int t = blockIdx.x * blockDim.x + threadIdx.x;
    if (t >= TE * HEADS) return;
    int e = t >> 3, hd = t & 7;
    int s, d;
    if (e < E) { s = src[e]; d = dst[e]; } else { s = d = e - E; }
    int dh = d * HEADS + hd;
    float v = lrelu(als[s * HEADS + hd] + ald[dh]);
    float alpha = __expf(v - mbuf[dh]) / (sbuf[dh] + 1e-16f);
    const float* hs = h + (long)s * Cout_ + hd * C;
    float* od = obuf + (long)d * Cout_ + hd * C;
    for (int c = 0; c < C; ++c) atomicAdd(&od[c], hs[c] * alpha);
}

// ---------- BN stats: one block per column ----------
__global__ void bn_stats(const float* __restrict__ o, const float* __restrict__ bias,
                         float* __restrict__ mean, float* __restrict__ var, int n, int C) {
    int c = blockIdx.x;
    float b = bias[c];
    float s = 0.f, s2 = 0.f;
    for (int i = threadIdx.x; i < n; i += blockDim.x) {
        float v = o[(long)i * C + c] + b;
        s += v; s2 += v * v;
    }
    __shared__ float shs[256], shq[256];
    int tid = threadIdx.x;
    shs[tid] = s; shq[tid] = s2;
    __syncthreads();
    for (int off = 128; off > 0; off >>= 1) {
        if (tid < off) { shs[tid] += shs[tid + off]; shq[tid] += shq[tid + off]; }
        __syncthreads();
    }
    if (tid == 0) {
        float mu = shs[0] / n;
        mean[c] = mu;
        var[c] = shq[0] / n - mu * mu;
    }
}

// ---------- fused bias + BN + ReLU; emit bf16 next-layer input and/or f32 ----------
__global__ void bn_relu(const float* __restrict__ o, const float* __restrict__ bias,
                        const float* __restrict__ mean, const float* __restrict__ var,
                        const float* __restrict__ g, const float* __restrict__ be,
                        unsigned short* __restrict__ xb16, float* __restrict__ y32,
                        int n, int C) {
    int t = blockIdx.x * blockDim.x + threadIdx.x;
    if (t >= n * C) return;
    int c = t % C;
    float v = o[t] + bias[c];
    float y = (v - mean[c]) * rsqrtf(var[c] + EPS_BN) * g[c] + be[c];
    y = fmaxf(y, 0.f);
    if (xb16) xb16[t] = f2bf(y);
    if (y32)  y32[t] = y;
}

// ---------- global max pool ----------
__global__ void pool_max(const float* __restrict__ y, const int* __restrict__ batch,
                         float* __restrict__ pooled, int n, int C) {
    int t = blockIdx.x * blockDim.x + threadIdx.x;
    if (t >= n * C) return;
    int node = t / C, c = t % C;
    atomicMaxF(&pooled[(long)batch[node] * C + c], y[t]);
}

// ---------- final FC: out[B,10] = pooled[B,512] @ fcW[512,10] + fcb ----------
__global__ void fc_kernel(const float* __restrict__ pooled, const float* __restrict__ W,
                          const float* __restrict__ b, float* __restrict__ out,
                          int Brows, int K, int O) {
    int t = blockIdx.x * blockDim.x + threadIdx.x;
    if (t >= Brows * O) return;
    int r = t / O, c = t % O;
    float s = b[c];
    for (int k = 0; k < K; ++k) s += pooled[(long)r * K + k] * W[(long)k * O + c];
    out[t] = s;
}

extern "C" void kernel_launch(void* const* d_in, const int* in_sizes, int n_in,
                              void* d_out, int out_size, void* d_ws, size_t ws_size,
                              hipStream_t stream) {
    (void)n_in; (void)ws_size;
    const int N  = in_sizes[0] / 3;
    const int E  = in_sizes[1] / 2;
    const int B  = out_size / 10;
    const int TE = E + N;

    const float* x     = (const float*)d_in[0];
    const int*   src   = (const int*)d_in[1];
    const int*   dst   = src + E;
    const int*   batch = (const int*)d_in[2];
    const float* Wp[3]  = {(const float*)d_in[3],  (const float*)d_in[9],  (const float*)d_in[15]};
    const float* asp[3] = {(const float*)d_in[4],  (const float*)d_in[10], (const float*)d_in[16]};
    const float* adp[3] = {(const float*)d_in[5],  (const float*)d_in[11], (const float*)d_in[17]};
    const float* bp[3]  = {(const float*)d_in[6],  (const float*)d_in[12], (const float*)d_in[18]};
    const float* gp[3]  = {(const float*)d_in[7],  (const float*)d_in[13], (const float*)d_in[19]};
    const float* bep[3] = {(const float*)d_in[8],  (const float*)d_in[14], (const float*)d_in[20]};
    const float* fcW = (const float*)d_in[21];
    const float* fcb = (const float*)d_in[22];
    float* out = (float*)d_out;

    const int Kin[3]  = {3, 128, 256};
    const int Kpad[3] = {32, 128, 256};
    const int Cout[3] = {128, 256, 512};
    const int Cph[3]  = {16, 32, 64};

    // workspace carve (everything fits well inside the 192 MB L2: ~95 MB total)
    char* w = (char*)d_ws;
    auto carve = [&](size_t bytes) -> void* {
        void* p = (void*)w;
        w += (bytes + 255) & ~(size_t)255;
        return p;
    };
    unsigned short* xb   = (unsigned short*)carve((size_t)N * 256 * 2);   // bf16 activations
    unsigned short* wb   = (unsigned short*)carve((size_t)256 * 512 * 2); // bf16 packed weights
    float* hbuf   = (float*)carve((size_t)N * 512 * 4);   // pre-attention h / final y
    float* obuf   = (float*)carve((size_t)N * 512 * 4);   // attention output
    float* als    = (float*)carve((size_t)N * HEADS * 4);
    float* ald    = (float*)carve((size_t)N * HEADS * 4);
    float* mbuf   = (float*)carve((size_t)N * HEADS * 4);
    float* sbuf   = (float*)carve((size_t)N * HEADS * 4);
    float* mean   = (float*)carve(512 * 4);
    float* var    = (float*)carve(512 * 4);
    float* pooled = (float*)carve((size_t)B * 512 * 4);

    auto nb = [](long t, int b) { return (int)((t + b - 1) / b); };

    for (int L = 0; L < 3; ++L) {
        // pack weights to bf16 WMMA B-fragment order (zero-padded K rows)
        pack_b_frag<<<nb((long)Kpad[L] * Cout[L], 256), 256, 0, stream>>>(
            Wp[L], wb, Kpad[L], Cout[L], Kin[L]);
        if (L == 0) {
            // x [N,3] -> bf16 [N,32] zero-padded
            cast_pad_bf16<<<nb((long)N * 32, 256), 256, 0, stream>>>(x, xb, N, 32, N, 3);
        }
        // h = xb @ W  (bf16 WMMA, f32 accumulate); wave computes 16x64 strip
        {
            int waves = (N / 16) * (Cout[L] / 64);
            gat_gemm_wmma<<<nb(waves, 8), 256, 0, stream>>>(xb, wb, hbuf, N, Kpad[L], Cout[L]);
        }
        attn_logits<<<nb((long)N * HEADS, 256), 256, 0, stream>>>(hbuf, asp[L], adp[L], als, ald, N, Cph[L]);
        fill_f32<<<nb((long)N * HEADS, 256), 256, 0, stream>>>(mbuf, N * HEADS, -INFINITY);
        fill_f32<<<nb((long)N * HEADS, 256), 256, 0, stream>>>(sbuf, N * HEADS, 0.f);
        fill_f32<<<nb((long)N * Cout[L], 256), 256, 0, stream>>>(obuf, N * Cout[L], 0.f);
        edge_max<<<nb((long)TE * HEADS, 256), 256, 0, stream>>>(src, dst, E, TE, als, ald, mbuf);
        edge_sum<<<nb((long)TE * HEADS, 256), 256, 0, stream>>>(src, dst, E, TE, als, ald, mbuf, sbuf);
        edge_scatter<<<nb((long)TE * HEADS, 256), 256, 0, stream>>>(src, dst, E, TE, als, ald,
                                                                    mbuf, sbuf, hbuf, obuf,
                                                                    Cout[L], Cph[L]);
        bn_stats<<<Cout[L], 256, 0, stream>>>(obuf, bp[L], mean, var, N, Cout[L]);
        bn_relu<<<nb((long)N * Cout[L], 256), 256, 0, stream>>>(
            obuf, bp[L], mean, var, gp[L], bep[L],
            (L < 2) ? xb : (unsigned short*)nullptr,
            (L == 2) ? hbuf : (float*)nullptr,
            N, Cout[L]);
    }

    fill_f32<<<nb((long)B * 512, 256), 256, 0, stream>>>(pooled, B * 512, -INFINITY);
    pool_max<<<nb((long)N * 512, 256), 256, 0, stream>>>(hbuf, batch, pooled, N, 512);
    fc_kernel<<<nb((long)B * 10, 64), 64, 0, stream>>>(pooled, fcW, fcb, out, B, 512, 10);
}
